// GraphGenerator_31628139168415
// MI455X (gfx1250) — matrix operand order repaired and checked by reading, not compiled, not run on hardware
//
#include <hip/hip_runtime.h>
#include <hip/hip_bf16.h>
#include <math.h>
#include <stdint.h>

#define NN 50000
#define NE 500000
#define FD 128

typedef _Float16 h8   __attribute__((ext_vector_type(8)));
typedef _Float16 v16h __attribute__((ext_vector_type(16)));
typedef float    v8f  __attribute__((ext_vector_type(8)));
typedef int      v4i  __attribute__((ext_vector_type(4)));

union AF { v16h v; h8 h[2]; };

// ---------- gfx1250 async global->LDS staging (guarded; falls back to ds stores) ----------
typedef __attribute__((address_space(1))) v4i as1_v4i;
typedef __attribute__((address_space(3))) v4i as3_v4i;

#if __has_builtin(__builtin_amdgcn_global_load_async_to_lds_b128)
#define HAVE_ASYNC_LDS 1
#else
#define HAVE_ASYNC_LDS 0
#endif

__device__ __forceinline__ void stage16(void* l, const void* g) {
#if HAVE_ASYNC_LDS
    __builtin_amdgcn_global_load_async_to_lds_b128((as1_v4i*)g, (as3_v4i*)l, 0, 0);
#else
    *(h8*)l = *(const h8*)g;
#endif
}

__device__ __forceinline__ void stage_wait_barrier() {
#if HAVE_ASYNC_LDS
#if __has_builtin(__builtin_amdgcn_s_wait_asynccnt)
    __builtin_amdgcn_s_wait_asynccnt(0);
#else
    asm volatile("s_wait_asynccnt 0x0" ::: "memory");
#endif
#endif
    __syncthreads();
}

// ---------------- utility kernels ----------------

__global__ void fill_f32(float* p, float v, int n) {
    int i = blockIdx.x * 256 + threadIdx.x;
    if (i < n) p[i] = v;
}

__global__ void conv_f16(const float* __restrict__ in, _Float16* __restrict__ out, int n) {
    int i = blockIdx.x * 256 + threadIdx.x;
    if (i < n) out[i] = (_Float16)in[i];
}

// in: [R,C] row-major f32  ->  out: [C,R] row-major f16  (out[c][r] = in[r][c])
__global__ void transpose_w(const float* __restrict__ in, _Float16* __restrict__ out, int R, int C) {
    int i = blockIdx.x * 256 + threadIdx.x;
    if (i >= R * C) return;
    int r = i / C, c = i % C;
    out[(size_t)c * R + r] = (_Float16)in[i];
}

// ---------------- WMMA GEMM: out[M,128] = A16[M,128] @ W  (WT[n][k] pre-transposed) -------
// Block: 16 rows x 128 cols, 8 waves (one 16x16 col tile each). A tile staged in LDS
// (async-to-LDS), padded stride 136 halves (272B = 68 words, 68 mod 64 = 4 -> conflict-free).
template <bool BIAS, bool RELU, bool OUTF, bool OUT16>
__global__ void gemm128(const _Float16* __restrict__ A,
                        const _Float16* __restrict__ WT,
                        const float* __restrict__ bias,
                        float* __restrict__ outF,
                        _Float16* __restrict__ out16,
                        int M) {
    __shared__ _Float16 sA[16 * 136];
    int wave  = threadIdx.x >> 5;   // 0..7 -> output col tile
    int lane  = threadIdx.x & 31;
    int row0  = blockIdx.x * 16;
    int n0    = wave * 16;
    int lrow  = lane & 15;
    int khalf = lane >> 4;

    // stage A tile: thread t copies 16B chunk (row = t/16, chunk = t%16)
    {
        int r  = threadIdx.x >> 4;
        int ck = threadIdx.x & 15;
        stage16(sA + r * 136 + ck * 8, A + (size_t)(row0 + r) * FD + ck * 8);
    }
    stage_wait_barrier();

    v8f acc = {};
    const _Float16* arow = sA + lrow * 136;
#pragma unroll
    for (int kb = 0; kb < 128; kb += 32) {
        AF a, b;
        a.h[0] = *(const h8*)(arow + kb + khalf * 8);
        a.h[1] = *(const h8*)(arow + kb + 16 + khalf * 8);
        const _Float16* bcol = WT + (size_t)(n0 + lrow) * FD + kb + khalf * 8;
        b.h[0] = *(const h8*)(bcol);
        b.h[1] = *(const h8*)(bcol + 16);
        acc = __builtin_amdgcn_wmma_f32_16x16x32_f16(false, a.v, false, b.v,
                                                     (short)0, acc, false, false);
    }
    // D layout: lane L, vgpr r -> row = r + (L>=16 ? 8 : 0), col = n0 + (L&15)
    int col  = n0 + lrow;
    float bv = BIAS ? bias[col] : 0.f;
#pragma unroll
    for (int r = 0; r < 8; ++r) {
        float v = acc[r] + bv;
        if (RELU) v = v > 0.f ? v : 0.f;
        size_t o = (size_t)(row0 + r + khalf * 8) * FD + col;
        if (OUTF)  outF[o]  = v;
        if (OUT16) out16[o] = (_Float16)v;
    }
}

// ---------------- attention scalar kernels ----------------

__global__ void compute_sd(const float* __restrict__ h,
                           const float* __restrict__ as_, const float* __restrict__ ad_,
                           float* __restrict__ s, float* __restrict__ d, int N) {
    int wave = threadIdx.x >> 5, lane = threadIdx.x & 31;
    int node = blockIdx.x * 8 + wave;
    if (node >= N) return;
    float4 hv = ((const float4*)(h + (size_t)node * FD))[lane];
    float4 av = ((const float4*)as_)[lane];
    float4 dv = ((const float4*)ad_)[lane];
    float ss = hv.x * av.x + hv.y * av.y + hv.z * av.z + hv.w * av.w;
    float dd = hv.x * dv.x + hv.y * dv.y + hv.z * dv.z + hv.w * dv.w;
#pragma unroll
    for (int off = 16; off; off >>= 1) {
        ss += __shfl_xor(ss, off, 32);
        dd += __shfl_xor(dd, off, 32);
    }
    if (lane == 0) { s[node] = ss; d[node] = dd; }
}

__device__ __forceinline__ float leaky(float x) { return x >= 0.f ? x : 0.2f * x; }

__global__ void edge_max(const float* __restrict__ s, const float* __restrict__ d,
                         const int* __restrict__ src, const int* __restrict__ dst,
                         float* __restrict__ m, int E, int N) {
    int i = blockIdx.x * 256 + threadIdx.x;
    if (i >= E + N) return;
    int si = (i < E) ? src[i] : (i - E);
    int di = (i < E) ? dst[i] : (i - E);
    float e = leaky(s[si] + d[di]);
    // float atomic max via signed/unsigned int trick (m initialized to -inf)
    if (e >= 0.f) atomicMax((int*)(m + di), __float_as_int(e));
    else          atomicMin((unsigned int*)(m + di), __float_as_uint(e));
}

__global__ void edge_exp(const float* __restrict__ s, const float* __restrict__ d,
                         const int* __restrict__ src, const int* __restrict__ dst,
                         const float* __restrict__ m, float* __restrict__ ex,
                         float* __restrict__ den, int E, int N) {
    int i = blockIdx.x * 256 + threadIdx.x;
    if (i >= E + N) return;
    int si = (i < E) ? src[i] : (i - E);
    int di = (i < E) ? dst[i] : (i - E);
    float e = leaky(s[si] + d[di]);
    float v = __expf(e - m[di]);
    ex[i] = v;
    atomicAdd(den + di, v);
}

__global__ void aggregate(const float* __restrict__ h,
                          const int* __restrict__ src, const int* __restrict__ dst,
                          const float* __restrict__ ex, const float* __restrict__ den,
                          float* __restrict__ out, int E, int N) {
    int wave = threadIdx.x >> 5, lane = threadIdx.x & 31;
    int i = blockIdx.x * 8 + wave;   // edge id (incl. self loops)
    if (i >= E + N) return;
    int si = (i < E) ? src[i] : (i - E);
    int di = (i < E) ? dst[i] : (i - E);
    float w = ex[i] / (den[di] + 1e-16f);
    float4 hv = ((const float4*)(h + (size_t)si * FD))[lane];
    float* op = out + (size_t)di * FD + lane * 4;
    atomicAdd(op + 0, w * hv.x);
    atomicAdd(op + 1, w * hv.y);
    atomicAdd(op + 2, w * hv.z);
    atomicAdd(op + 3, w * hv.w);
}

__global__ void bias_relu_conv(float* __restrict__ z, const float* __restrict__ b,
                               _Float16* __restrict__ z16, int M) {
    int i = blockIdx.x * 256 + threadIdx.x;
    if (i >= M * FD) return;
    float v = z[i] + b[i & (FD - 1)];
    v = v > 0.f ? v : 0.f;
    z[i] = v;
    z16[i] = (_Float16)v;
}

// ---------------- fused edge-predictor MLP (WMMA + LDS-resident weights) ----------------
// Per wave: 16 edges, K=256 over concat(z[src],z[dst]), 8 col tiles, then 128->1 dot+sigmoid.
// We1^T (64KB) staged to LDS chunk-major: chunk c=k/16, laddr = (c*128 + row)*32B.
__global__ void edge_mlp(const _Float16* __restrict__ z16,
                         const int* __restrict__ src, const int* __restrict__ dst,
                         const _Float16* __restrict__ WTe1,  // [128][256]
                         const float* __restrict__ be1,
                         const float* __restrict__ We2,      // [128]
                         const float* __restrict__ be2,
                         float* __restrict__ outp, int E) {
    __shared__ _Float16 sW[128 * 256];  // 64KB
    // stage: 2048 chunks of 16 halves; thread handles 8 chunks (cid = c*128 + row)
#pragma unroll
    for (int j = 0; j < 8; ++j) {
        int cid = threadIdx.x + j * 256;
        int c = cid >> 7, row = cid & 127;
        const _Float16* g = WTe1 + (size_t)row * 256 + c * 16;
        _Float16* l = sW + (size_t)cid * 16;
        stage16(l, g);
        stage16(l + 8, g + 8);
    }
    stage_wait_barrier();

    int wave = threadIdx.x >> 5;
    int lane = threadIdx.x & 31;
    int base = (blockIdx.x * 8 + wave) * 16;
    if (base >= E) return;                 // uniform per wave (after barrier)
    int lrow  = lane & 15;
    int khalf = lane >> 4;
    int sidx = src[base + lrow];
    int didx = dst[base + lrow];
    const _Float16* zs = z16 + (size_t)sidx * FD;
    const _Float16* zd = z16 + (size_t)didx * FD;

    v8f acc[8];
    v8f zero = {};
#pragma unroll
    for (int t = 0; t < 8; ++t) acc[t] = zero;

#pragma unroll
    for (int kb = 0; kb < 256; kb += 32) {
        int K1 = kb + khalf * 16;
        int K2 = kb + khalf * 16 + 8;
        AF a;
        a.h[0] = *(const h8*)((K1 < FD) ? (zs + K1) : (zd + K1 - FD));
        a.h[1] = *(const h8*)((K2 < FD) ? (zs + K2) : (zd + K2 - FD));
        int cidx = (kb >> 4) + khalf;      // 16-half chunk index
#pragma unroll
        for (int t = 0; t < 8; ++t) {
            AF b;
            const _Float16* bp = sW + ((size_t)(cidx << 7) + t * 16 + lrow) * 16;
            b.h[0] = *(const h8*)(bp);
            b.h[1] = *(const h8*)(bp + 8);
            acc[t] = __builtin_amdgcn_wmma_f32_16x16x32_f16(false, a.v, false, b.v,
                                                            (short)0, acc[t], false, false);
        }
    }
    // second layer: p = sigmoid( relu(g + be1) . We2 + be2 )
    float w2[8], bv[8];
#pragma unroll
    for (int t = 0; t < 8; ++t) { w2[t] = We2[t * 16 + lrow]; bv[t] = be1[t * 16 + lrow]; }
    float b2 = be2[0];
#pragma unroll
    for (int r = 0; r < 8; ++r) {
        float val = 0.f;
#pragma unroll
        for (int t = 0; t < 8; ++t) {
            float g = acc[t][r] + bv[t];
            g = g > 0.f ? g : 0.f;
            val += g * w2[t];
        }
        val += __shfl_xor(val, 8, 32);
        val += __shfl_xor(val, 4, 32);
        val += __shfl_xor(val, 2, 32);
        val += __shfl_xor(val, 1, 32);
        if (lrow == 0) {
            int edge = base + r + khalf * 8;
            outp[edge] = 1.f / (1.f + __expf(-(val + b2)));
        }
    }
}

// ---------------- host ----------------

extern "C" void kernel_launch(void* const* d_in, const int* in_sizes, int n_in,
                              void* d_out, int out_size, void* d_ws, size_t ws_size,
                              hipStream_t stream) {
    const float* x   = (const float*)d_in[0];
    const int*   ei  = (const int*)  d_in[1];
    const float* W1  = (const float*)d_in[2];
    const float* a1s = (const float*)d_in[3];
    const float* a1d = (const float*)d_in[4];
    const float* b1  = (const float*)d_in[5];
    const float* W2  = (const float*)d_in[6];
    const float* a2s = (const float*)d_in[7];
    const float* a2d = (const float*)d_in[8];
    const float* b2  = (const float*)d_in[9];
    const float* W3  = (const float*)d_in[10];
    const float* a3s = (const float*)d_in[11];
    const float* a3d = (const float*)d_in[12];
    const float* b3  = (const float*)d_in[13];
    const float* Wn1 = (const float*)d_in[14];
    const float* bn1 = (const float*)d_in[15];
    const float* Wn2 = (const float*)d_in[16];
    const float* bn2 = (const float*)d_in[17];
    const float* We1 = (const float*)d_in[18];
    const float* be1 = (const float*)d_in[19];
    const float* We2 = (const float*)d_in[20];
    const float* be2 = (const float*)d_in[21];
    (void)in_sizes; (void)n_in; (void)out_size; (void)ws_size;

    const int* src = ei;
    const int* dst = ei + NE;

    char* ws = (char*)d_ws;
    size_t off = 0;
    auto carve = [&](size_t bytes) -> char* {
        char* p = ws + off;
        off = (off + bytes + 255) & ~(size_t)255;
        return p;
    };
    _Float16* x16  = (_Float16*)carve((size_t)NN * FD * 2);
    _Float16* z16  = (_Float16*)carve((size_t)NN * FD * 2);
    _Float16* t16  = (_Float16*)carve((size_t)NN * FD * 2);
    float*    h    = (float*)   carve((size_t)NN * FD * 4);
    float*    z    = (float*)   carve((size_t)NN * FD * 4);
    float*    sb   = (float*)   carve((size_t)NN * 4);
    float*    db   = (float*)   carve((size_t)NN * 4);
    float*    mb   = (float*)   carve((size_t)NN * 4);
    float*    den  = (float*)   carve((size_t)NN * 4);
    float*    ex   = (float*)   carve((size_t)(NE + NN) * 4);
    _Float16* WT1  = (_Float16*)carve(FD * FD * 2);
    _Float16* WT2  = (_Float16*)carve(FD * FD * 2);
    _Float16* WT3  = (_Float16*)carve(FD * FD * 2);
    _Float16* WTn1 = (_Float16*)carve(FD * FD * 2);
    _Float16* WTn2 = (_Float16*)carve(FD * FD * 2);
    _Float16* WTe1 = (_Float16*)carve(256 * FD * 2);

    // weight transposes (f32 -> f16, [K,N] -> [N,K])
    transpose_w<<<64, 256, 0, stream>>>(W1,  WT1,  128, 128);
    transpose_w<<<64, 256, 0, stream>>>(W2,  WT2,  128, 128);
    transpose_w<<<64, 256, 0, stream>>>(W3,  WT3,  128, 128);
    transpose_w<<<64, 256, 0, stream>>>(Wn1, WTn1, 128, 128);
    transpose_w<<<64, 256, 0, stream>>>(Wn2, WTn2, 128, 128);
    transpose_w<<<128, 256, 0, stream>>>(We1, WTe1, 256, 128);

    conv_f16<<<(NN * FD) / 256, 256, 0, stream>>>(x, x16, NN * FD);

    const int Etot = NE + NN;
    auto gat_layer = [&](const _Float16* in16, const _Float16* WT,
                         const float* as_, const float* ad_, const float* bb) {
        gemm128<false, false, true, false>
            <<<NN / 16, 256, 0, stream>>>(in16, WT, nullptr, h, nullptr, NN);
        compute_sd<<<(NN + 7) / 8, 256, 0, stream>>>(h, as_, ad_, sb, db, NN);
        fill_f32<<<(NN + 255) / 256, 256, 0, stream>>>(mb, -INFINITY, NN);
        fill_f32<<<(NN + 255) / 256, 256, 0, stream>>>(den, 0.f, NN);
        fill_f32<<<(NN * FD) / 256, 256, 0, stream>>>(z, 0.f, NN * FD);
        edge_max<<<(Etot + 255) / 256, 256, 0, stream>>>(sb, db, src, dst, mb, NE, NN);
        edge_exp<<<(Etot + 255) / 256, 256, 0, stream>>>(sb, db, src, dst, mb, ex, den, NE, NN);
        aggregate<<<(Etot + 7) / 8, 256, 0, stream>>>(h, src, dst, ex, den, z, NE, NN);
        bias_relu_conv<<<(NN * FD) / 256, 256, 0, stream>>>(z, bb, z16, NN);
    };
    gat_layer(x16, WT1, a1s, a1d, b1);
    gat_layer(z16, WT2, a2s, a2d, b2);
    gat_layer(z16, WT3, a3s, a3d, b3);

    // node predictor MLP: relu(z@Wn1+bn1)@Wn2+bn2 -> d_out[0 : NN*FD]
    gemm128<true, true, false, true>
        <<<NN / 16, 256, 0, stream>>>(z16, WTn1, bn1, nullptr, t16, NN);
    gemm128<true, false, true, false>
        <<<NN / 16, 256, 0, stream>>>(t16, WTn2, bn2, (float*)d_out, nullptr, NN);

    // edge predictor -> d_out[NN*FD : NN*FD + NE]
    int nwaves = NE / 16;  // 31250
    edge_mlp<<<(nwaves + 7) / 8, 256, 0, stream>>>(z16, src, dst, WTe1, be1, We2, be2,
                                                   (float*)d_out + (size_t)NN * FD, NE);
}
